// OphFusionNet_70308614635684
// MI455X (gfx1250) — matrix operand order repaired and checked by reading, not compiled, not run on hardware
//
#include <hip/hip_runtime.h>
#include <hip/hip_fp16.h>

typedef __attribute__((ext_vector_type(16))) _Float16 v16h;
typedef __attribute__((ext_vector_type(8)))  _Float16 v8h;
typedef __attribute__((ext_vector_type(8)))  float    v8f;

// ---------------------------------------------------------------------------
// WMMA operand loaders for C = X * Y^T, X (MxK) row-major, Y (NxK) row-major,
// both f16.  Contiguous k order, not the ISA table order (16-bit A 16x32, B 32x16, wave32).
// A: lane L (m=L&15, hi=L>>4) holds K = {16*hi+0..15} of row m.
// B: lane L (n=L&15, hi=L>>4) holds K = {16*hi+0..15} of row n of Y.
// ---------------------------------------------------------------------------
__device__ __forceinline__ v16h wmma_load_a(const _Float16* X, int ld, int row0, int k0) {
  int lane = threadIdx.x & 31;
  int m = lane & 15, hi = lane >> 4;
  const _Float16* p = X + (size_t)(row0 + m) * ld + k0 + 16 * hi;
  v16h r = *(const v16h*)(p);
  return r;
}

__device__ __forceinline__ v16h wmma_load_bt(const _Float16* Y, int ld, int n0, int k0) {
  int lane = threadIdx.x & 31;
  int n = lane & 15, hi = lane >> 4;
  return *(const v16h*)(Y + (size_t)(n0 + n) * ld + k0 + 16 * hi);
}

__device__ __forceinline__ v8f wmma16(v16h a, v16h b, v8f c) {
  return __builtin_amdgcn_wmma_f32_16x16x32_f16(false, a, false, b, (short)0, c, false, false);
}

// ---------------------------------------------------------------------------
// Tensor Data Mover: 2D tile (rows x cols f16 elements, row stride in elems)
// global -> LDS.  D# per CDNA5 ISA ch.8 (group0/group1; groups 2/3 zero = 2D).
// This toolchain's builtin is the 6-arg clang-23 form:
//   (u32x4 g0, i32x8 g1, i32x4 g2, i32x4 g3, i32x8 g4, i32 cpol)
// ---------------------------------------------------------------------------
#if __has_builtin(__builtin_amdgcn_tensor_load_to_lds) && \
    __has_builtin(__builtin_amdgcn_s_wait_tensorcnt)
#define HAVE_TDM 1
typedef __attribute__((ext_vector_type(4))) unsigned int u32x4;
typedef __attribute__((ext_vector_type(8))) int i32x8;
typedef __attribute__((ext_vector_type(4))) int i32x4;

__device__ __forceinline__ void tdm_load_2d_f16(unsigned lds_off, const _Float16* gptr,
                                                unsigned rows, unsigned cols,
                                                unsigned stride_elems) {
  unsigned long long ga = (unsigned long long)(size_t)gptr;
  u32x4 g0;
  g0[0] = 1u;                                            // count=1, user mode
  g0[1] = lds_off;                                       // lds_addr (bytes)
  g0[2] = (unsigned)(ga & 0xFFFFFFFFu);                  // global_addr[31:0]
  g0[3] = (unsigned)((ga >> 32) & 0x01FFFFFFu) | (2u << 30);  // addr[56:32], type=2
  i32x8 g1;
  g1[0] = (int)(1u << 16);                               // data_size=1 (2 bytes)
  g1[1] = (int)((cols & 0xFFFFu) << 16);                 // tensor_dim0[15:0]
  g1[2] = (int)((cols >> 16) | ((rows & 0xFFFFu) << 16));// tensor_dim0[31:16], dim1[15:0]
  g1[3] = (int)((rows >> 16) | (cols << 16));            // dim1[31:16], tile_dim0
  g1[4] = (int)rows;                                     // tile_dim1 (tile_dim2 = 0)
  g1[5] = (int)stride_elems;                             // tensor_dim0_stride[31:0]
  g1[6] = 0;
  g1[7] = 0;
  i32x4 z4 = {0, 0, 0, 0};
  i32x8 z8 = {0, 0, 0, 0, 0, 0, 0, 0};
  __builtin_amdgcn_tensor_load_to_lds(g0, g1, z4, z4, z8, 0);
}
#endif

// ---------------------------------------------------------------------------
// Kernel 1: convert weights to f16.  Wall = [Wq ; Wkv] (3072x1024), Wp (1024x1024)
// ---------------------------------------------------------------------------
__global__ void __launch_bounds__(256) k_conv_weights(const float* __restrict__ Wq,
                                                      const float* __restrict__ Wkv,
                                                      const float* __restrict__ Wproj,
                                                      _Float16* __restrict__ Wall,
                                                      _Float16* __restrict__ Wp) {
  const int nAll = 3072 * 1024, nQ = 1024 * 1024, nP = 1024 * 1024;
  for (int idx = blockIdx.x * blockDim.x + threadIdx.x; idx < nAll + nP;
       idx += gridDim.x * blockDim.x) {
    if (idx < nAll) {
      float v = (idx < nQ) ? Wq[idx] : Wkv[idx - nQ];
      Wall[idx] = (_Float16)v;
    } else {
      Wp[idx - nAll] = (_Float16)Wproj[idx - nAll];
    }
  }
}

// ---------------------------------------------------------------------------
// Kernel 2: x (B,1024,49) f32 -> xt (B,64,1024) f16, token-major, rows 49..63 = 0
// ---------------------------------------------------------------------------
__global__ void __launch_bounds__(256) k_prep_xt(const float* __restrict__ x,
                                                 _Float16* __restrict__ xt) {
  int cb = blockIdx.x, b = blockIdx.y;
  __shared__ float sX[128 * 50];
  const float* xp = x + ((size_t)b * 1024 + cb * 128) * 49;
  for (int idx = threadIdx.x; idx < 128 * 49; idx += blockDim.x) {
    int c = idx / 49, n = idx % 49;
    sX[c * 50 + n] = xp[idx];
  }
  __syncthreads();
  _Float16* op = xt + (size_t)b * 64 * 1024 + cb * 128;
  for (int idx = threadIdx.x; idx < 64 * 128; idx += blockDim.x) {
    int n = idx >> 7, c = idx & 127;
    float v = (n < 49) ? sX[c * 50 + n] : 0.0f;
    op[(size_t)n * 1024 + c] = (_Float16)v;
  }
}

// ---------------------------------------------------------------------------
// Kernel 3: QKV GEMM.  block (fb 0..23, b): 64 tokens x 128-feature band.
// 4 waves, each: full M=64 (4 m-tiles) x 32 features (2 n-tiles), K=1024.
// 8 WMMAs per 6 b128-load-pairs per k-step.
// ---------------------------------------------------------------------------
__global__ void __launch_bounds__(128) k_qkv_gemm(const _Float16* __restrict__ xt,
                                                  const _Float16* __restrict__ Wall,
                                                  const float* __restrict__ bq,
                                                  const float* __restrict__ bkv,
                                                  _Float16* __restrict__ qkv) {
  int fb = blockIdx.x, b = blockIdx.y;
  int wave = threadIdx.x >> 5, lane = threadIdx.x & 31;
  const _Float16* A = xt + (size_t)b * 64 * 1024;
  int n0 = fb * 128 + wave * 32;
  v8f acc[4][2] = {};
  for (int k0 = 0; k0 < 1024; k0 += 32) {
    v16h a[4];
#pragma unroll
    for (int mt = 0; mt < 4; ++mt) a[mt] = wmma_load_a(A, 1024, mt * 16, k0);
    v16h b0 = wmma_load_bt(Wall, 1024, n0, k0);
    v16h b1 = wmma_load_bt(Wall, 1024, n0 + 16, k0);
#pragma unroll
    for (int mt = 0; mt < 4; ++mt) {
      acc[mt][0] = wmma16(a[mt], b0, acc[mt][0]);
      acc[mt][1] = wmma16(a[mt], b1, acc[mt][1]);
    }
  }
  int nloc = lane & 15, hi = lane >> 4;
#pragma unroll
  for (int mt = 0; mt < 4; ++mt)
#pragma unroll
    for (int nt = 0; nt < 2; ++nt) {
      int c = n0 + nt * 16 + nloc;                       // feature 0..3071
      float bias = (c < 1024) ? bq[c] : bkv[c - 1024];
      float scl = (c < 1024) ? 0.08838834764831845f : 1.0f;  // D^-0.5, q only
#pragma unroll
      for (int r = 0; r < 8; ++r) {
        int m = mt * 16 + r + 8 * hi;
        if (m < 49) {
          float v = (acc[mt][nt][r] + bias) * scl;
          qkv[((size_t)b * 49 + m) * 3072 + c] = (_Float16)v;
        }
      }
    }
}

// ---------------------------------------------------------------------------
// Kernel 4: attention for one (b,h).  4 waves.  q/k/v staged by TDM.
// ---------------------------------------------------------------------------
__global__ void __launch_bounds__(128) k_attn(const _Float16* __restrict__ qkv,
                                              const float* __restrict__ rpb_table,
                                              const float* __restrict__ w_mix,
                                              _Float16* __restrict__ attn_out) {
  int bh = blockIdx.x;
  int b = bh >> 3, h = bh & 7;
  int tid = threadIdx.x, wave = tid >> 5, lane = tid & 31;

  __shared__ __align__(32) _Float16 sK[64 * 128];    // 16 KB
  __shared__ __align__(32) _Float16 sVT[128 * 64];   // 16 KB  (v^T: [d][token])
  __shared__ __align__(32) _Float16 sQbuf[64 * 128]; // 16 KB  (aliased by sA later)
  __shared__ __align__(32) float    sAttn[64 * 64];  // 16 KB  (aliased by v staging first)
  _Float16* sQ = sQbuf;
  _Float16* sA = sQbuf;               // reused after GEMM1 barrier
  _Float16* sVstage = (_Float16*)sAttn;  // v rows staged here before transpose

  const _Float16* qbase = qkv + (size_t)b * 49 * 3072 + h * 128;

#if defined(HAVE_TDM)
  if (wave == 0) {
    tdm_load_2d_f16((unsigned)(size_t)sQ,      qbase,        49u, 128u, 3072u);
    tdm_load_2d_f16((unsigned)(size_t)sK,      qbase + 1024, 49u, 128u, 3072u);
    tdm_load_2d_f16((unsigned)(size_t)sVstage, qbase + 2048, 49u, 128u, 3072u);
  }
  // zero pad rows (disjoint from TDM destinations) while DMA runs
  for (int idx = tid; idx < 15 * 128; idx += 128) {
    int row = 49 + (idx >> 7), col = idx & 127;
    sQ[row * 128 + col] = (_Float16)0.0f;
    sK[row * 128 + col] = (_Float16)0.0f;
    sVT[col * 64 + row] = (_Float16)0.0f;
  }
  if (wave == 0) __builtin_amdgcn_s_wait_tensorcnt(0);
  __syncthreads();
  // LDS -> LDS transpose of v (TDM cannot transpose)
  for (int idx = tid; idx < 49 * 128; idx += 128) {
    int row = idx >> 7, col = idx & 127;
    sVT[col * 64 + row] = sVstage[idx];
  }
  __syncthreads();   // sVstage (alias of sAttn) dead from here on
#else
  for (int idx = tid; idx < 64 * 128; idx += 128) {
    int row = idx >> 7, col = idx & 127;
    _Float16 qv = (_Float16)0.0f, kv = (_Float16)0.0f, vv = (_Float16)0.0f;
    if (row < 49) {
      size_t base = (size_t)row * 3072 + col;
      qv = qbase[base];
      kv = qbase[base + 1024];
      vv = qbase[base + 2048];
    }
    sQ[idx] = qv;
    sK[idx] = kv;
    sVT[col * 64 + row] = vv;
  }
  __syncthreads();
#endif

  // ---- GEMM1: attn = q . k^T  (64x64, K=128) ----
  {
    int m0 = wave * 16;
    v8f acc[4] = {};
#pragma unroll
    for (int k0 = 0; k0 < 128; k0 += 32) {
      v16h a = wmma_load_a(sQ, 128, m0, k0);
#pragma unroll
      for (int t = 0; t < 4; ++t) {
        v16h bop = wmma_load_bt(sK, 128, t * 16, k0);
        acc[t] = wmma16(a, bop, acc[t]);
      }
    }
    int nloc = lane & 15, hi = lane >> 4;
#pragma unroll
    for (int t = 0; t < 4; ++t)
#pragma unroll
      for (int r = 0; r < 8; ++r)
        sAttn[(m0 + r + 8 * hi) * 64 + t * 16 + nloc] = acc[t][r];
  }
  __syncthreads();

  // ---- zero sA pad (sQ dead now), then bias + dual-branch softmax ----
  for (int idx = tid; idx < 64 * 64; idx += 128) sA[idx] = (_Float16)0.0f;
  float wa = w_mix[0], wb = w_mix[1];
  float wmx = fmaxf(wa, wb);
  float e0 = __expf(wa - wmx), e1 = __expf(wb - wmx);
  float w0 = e0 / (e0 + e1), w1 = e1 / (e0 + e1);
  __syncthreads();

  if (tid < 49) {
    int m = tid, mi = m / 7, mj = m % 7;
    float* row = &sAttn[m * 64];
    float mx = -3.4e38f;
    for (int n = 0; n < 49; ++n) {
      int ni = n / 7, nj = n % 7;
      int ridx = (mi - ni + 6) * 13 + (mj - nj + 6);   // REL_IDX[m][n]
      float v = row[n] + rpb_table[ridx * 8 + h];
      row[n] = v;
      mx = fmaxf(mx, v);
    }
    float s = 0.0f;
    for (int n = 0; n < 49; ++n) s += __expf(row[n] - mx);
    float inv = 1.0f / s;
    for (int n = 0; n < 49; ++n) {
      float v = row[n];
      float a0 = __expf(v - mx) * inv;
      float rl = fmaxf(v, 0.0f);
      sA[m * 64 + n] = (_Float16)(w0 * a0 + w1 * rl * rl);
    }
  }
  __syncthreads();

  // ---- GEMM2: out = attn . v  via attn(64x64) . sVT^T (K=64, N=128) ----
  {
    int m0 = wave * 16;
    v8f acc[8] = {};
#pragma unroll
    for (int k0 = 0; k0 < 64; k0 += 32) {
      v16h a = wmma_load_a(sA, 64, m0, k0);
#pragma unroll
      for (int t = 0; t < 8; ++t) {
        v16h bop = wmma_load_bt(sVT, 64, t * 16, k0);
        acc[t] = wmma16(a, bop, acc[t]);
      }
    }
    int nloc = lane & 15, hi = lane >> 4;
    _Float16* obase = attn_out + (size_t)b * 64 * 1024 + h * 128;
#pragma unroll
    for (int t = 0; t < 8; ++t)
#pragma unroll
      for (int r = 0; r < 8; ++r) {
        int m = m0 + r + 8 * hi;
        if (m < 49)
          obase[(size_t)m * 1024 + t * 16 + nloc] = (_Float16)acc[t][r];
      }
  }
}

// ---------------------------------------------------------------------------
// Kernel 5: proj GEMM + bias -> transposed (B,1024,49) f32 output.
// block (fb 0..7, b): 64 tokens x 128-channel band; waves as in k_qkv_gemm.
// ---------------------------------------------------------------------------
__global__ void __launch_bounds__(128) k_proj(const _Float16* __restrict__ attn_out,
                                              const _Float16* __restrict__ Wp,
                                              const float* __restrict__ bproj,
                                              float* __restrict__ out) {
  int fb = blockIdx.x, b = blockIdx.y;
  int tid = threadIdx.x, wave = tid >> 5, lane = tid & 31;
  __shared__ float sOut[64 * 129];   // stride 129: conflict-free transpose
  const _Float16* A = attn_out + (size_t)b * 64 * 1024;
  int n0 = fb * 128 + wave * 32;
  v8f acc[4][2] = {};
  for (int k0 = 0; k0 < 1024; k0 += 32) {
    v16h a[4];
#pragma unroll
    for (int mt = 0; mt < 4; ++mt) a[mt] = wmma_load_a(A, 1024, mt * 16, k0);
    v16h b0 = wmma_load_bt(Wp, 1024, n0, k0);
    v16h b1 = wmma_load_bt(Wp, 1024, n0 + 16, k0);
#pragma unroll
    for (int mt = 0; mt < 4; ++mt) {
      acc[mt][0] = wmma16(a[mt], b0, acc[mt][0]);
      acc[mt][1] = wmma16(a[mt], b1, acc[mt][1]);
    }
  }
  int nloc = lane & 15, hi = lane >> 4;
#pragma unroll
  for (int mt = 0; mt < 4; ++mt)
#pragma unroll
    for (int nt = 0; nt < 2; ++nt) {
      int cl = wave * 32 + nt * 16 + nloc;               // 0..127 within band
      float bias = bproj[fb * 128 + cl];
#pragma unroll
      for (int r = 0; r < 8; ++r)
        sOut[(mt * 16 + r + 8 * hi) * 129 + cl] = acc[mt][nt][r] + bias;
    }
  __syncthreads();
  float* obase = out + ((size_t)b * 1024 + fb * 128) * 49;
  for (int idx = tid; idx < 128 * 49; idx += 128) {
    int cl = idx / 49, n = idx % 49;
    obase[(size_t)cl * 49 + n] = sOut[n * 129 + cl];
  }
}

// ---------------------------------------------------------------------------
extern "C" void kernel_launch(void* const* d_in, const int* in_sizes, int n_in,
                              void* d_out, int out_size, void* d_ws, size_t ws_size,
                              hipStream_t stream) {
  const float* x     = (const float*)d_in[0];
  const float* Wq    = (const float*)d_in[1];
  const float* bq    = (const float*)d_in[2];
  const float* Wkv   = (const float*)d_in[3];
  const float* bkv   = (const float*)d_in[4];
  const float* rpb   = (const float*)d_in[5];
  const float* wmix  = (const float*)d_in[6];
  const float* Wproj = (const float*)d_in[7];
  const float* bproj = (const float*)d_in[8];
  float* out = (float*)d_out;

  char* ws = (char*)d_ws;
  size_t off = 0;
  _Float16* xt   = (_Float16*)(ws + off); off += (size_t)512 * 64 * 1024 * 2;   //  64 MB
  _Float16* Wall = (_Float16*)(ws + off); off += (size_t)3072 * 1024 * 2;       //   6 MB
  _Float16* Wp   = (_Float16*)(ws + off); off += (size_t)1024 * 1024 * 2;       //   2 MB
  _Float16* qkv  = (_Float16*)(ws + off); off += (size_t)512 * 49 * 3072 * 2;   // 147 MB
  _Float16* aout = (_Float16*)(ws + off); off += (size_t)512 * 64 * 1024 * 2;   //  64 MB

  k_conv_weights<<<dim3(4096), dim3(256), 0, stream>>>(Wq, Wkv, Wproj, Wall, Wp);
  k_prep_xt<<<dim3(8, 512), dim3(256), 0, stream>>>(x, xt);
  k_qkv_gemm<<<dim3(24, 512), dim3(128), 0, stream>>>(xt, Wall, bq, bkv, qkv);
  k_attn<<<dim3(4096), dim3(128), 0, stream>>>(qkv, rpb, wmix, aout);
  k_proj<<<dim3(8, 512), dim3(128), 0, stream>>>(aout, Wp, bproj, out);
}
